// _netG_17085379903521
// MI455X (gfx1250) — compile-verified, matrix operand order checked by reading, not run on hardware
//
#include <hip/hip_runtime.h>
#include <hip/hip_fp16.h>

typedef __attribute__((ext_vector_type(16))) _Float16 v16h;
typedef __attribute__((ext_vector_type(8)))  _Float16 v8h;
typedef __attribute__((ext_vector_type(8)))  float    v8f;
typedef int vsi4 __attribute__((vector_size(4 * sizeof(int))));

#define WG 256
#define TBM 128
#define TBN 128
#define TBK 32

#if defined(__has_builtin)
#  if __has_builtin(__builtin_amdgcn_global_load_async_to_lds_b128)
#    define HAVE_ASYNC_LDS 1
#  endif
#  if __has_builtin(__builtin_amdgcn_s_wait_asynccnt)
#    define HAVE_WAIT_ASYNC 1
#  endif
#endif

// 16-byte global -> LDS copy; async (ASYNCcnt-tracked) when the CDNA5 builtin
// is available, otherwise a plain load/store pair.
__device__ inline void copy16_g2l(_Float16* lds, const _Float16* g) {
#ifdef HAVE_ASYNC_LDS
  __builtin_amdgcn_global_load_async_to_lds_b128((vsi4*)g, (vsi4*)lds, 0, 0);
#else
  *(v8h*)lds = *(const v8h*)g;
#endif
}

__device__ inline void zero16_l(_Float16* lds) {
  v8h z = {};
  *(v8h*)lds = z;
}

__device__ inline void wait_async0() {
#ifdef HAVE_ASYNC_LDS
#  ifdef HAVE_WAIT_ASYNC
  __builtin_amdgcn_s_wait_asynccnt(0);
#  else
  asm volatile("s_wait_asynccnt 0" ::: "memory");
#  endif
#endif
}

// ---------------------------------------------------------------------------
// Uniform NT f16 WMMA GEMM:  D[M][Np] (f32) = A[M][K] * B[Np][ldb] (first K cols)
// A row-major f16 (lda == K), B position-major f16 with row stride ldb >= K.
// Tile 128x128x32, 8 waves (2x4), wave = 64x32 via 4x2 WMMA 16x16x32 tiles.
// Double-buffered LDS with async global->LDS staging.
// ---------------------------------------------------------------------------
__device__ inline v16h load_frag(const _Float16* p) {
  v8h lo = *(const v8h*)p;          // K = base .. base+7
  v8h hi = *(const v8h*)(p + 16);   // K = base+16 .. base+23
  return __builtin_shufflevector(lo, hi, 0,1,2,3,4,5,6,7,8,9,10,11,12,13,14,15);
}

__global__ __launch_bounds__(WG)
void gemm_f16_wmma(const _Float16* __restrict__ A, const _Float16* __restrict__ Bm,
                   float* __restrict__ D, int M, int Np, int K, int ldb)
{
  __shared__ _Float16 As[2][TBM][TBK + 8];
  __shared__ _Float16 Bs[2][TBN][TBK + 8];
  const int tid  = threadIdx.x;
  const int wave = tid >> 5;
  const int lane = tid & 31;
  const int wm   = (wave >> 2) * 64;   // wave row offset in tile
  const int wn   = (wave & 3) * 32;    // wave col offset in tile
  const int bm0  = blockIdx.y * TBM;
  const int bn0  = blockIdx.x * TBN;
  const int lrow = lane & 15;
  const int lkh  = (lane >> 4) * 8;    // K half select (0 or 8)

  v8f acc[4][2] = {};

  auto stage = [&](int buf, int k0) {
    // A tile: 128 rows x 32 K, 16B chunks (K always a multiple of 8 here)
    for (int c = tid; c < TBM * 4; c += WG) {
      int r = c >> 2, cc = (c & 3) * 8;
      int gr = bm0 + r, gc = k0 + cc;
      _Float16* dst = &As[buf][r][cc];
      if (gr < M && gc < K) copy16_g2l(dst, A + (size_t)gr * K + gc);
      else                  zero16_l(dst);
    }
    // B tile: 128 positions x 32 K
    for (int c = tid; c < TBN * 4; c += WG) {
      int r = c >> 2, cc = (c & 3) * 8;
      int gn = bn0 + r, gc = k0 + cc;
      _Float16* dst = &Bs[buf][r][cc];
      if (gn < Np && gc < K) copy16_g2l(dst, Bm + (size_t)gn * ldb + gc);
      else                   zero16_l(dst);
    }
  };

  stage(0, 0);
  wait_async0();
  __syncthreads();

  int cb = 0;
  for (int k0 = 0; k0 < K; k0 += TBK) {
    if (k0 + TBK < K) stage(cb ^ 1, k0 + TBK);   // overlap next slab with compute

    v16h af[4], bf[2];
#pragma unroll
    for (int i = 0; i < 4; ++i) af[i] = load_frag(&As[cb][wm + i * 16 + lrow][lkh]);
#pragma unroll
    for (int j = 0; j < 2; ++j) bf[j] = load_frag(&Bs[cb][wn + j * 16 + lrow][lkh]);

#pragma unroll
    for (int i = 0; i < 4; ++i)
#pragma unroll
      for (int j = 0; j < 2; ++j)
        acc[i][j] = __builtin_amdgcn_wmma_f32_16x16x32_f16(
            false, af[i], false, bf[j], (short)0, acc[i][j], false, false);

    wait_async0();       // next-slab async copies landed in other buffer
    __syncthreads();     // everyone done reading current buffer
    cb ^= 1;
  }

  // ---- store: lane holds col n=(lane&15), rows mh..mh+7 over 8 VGPRs ----
  const int cn = lane & 15;
  const int mh = (lane >> 4) * 8;
#pragma unroll
  for (int i = 0; i < 4; ++i)
#pragma unroll
    for (int j = 0; j < 2; ++j) {
      int gn = bn0 + wn + j * 16 + cn;
      if (gn >= Np) continue;
#pragma unroll
      for (int r = 0; r < 8; ++r) {
        int gm = bm0 + wm + i * 16 + mh + r;
        if (gm < M) D[(size_t)gm * Np + gn] = acc[i][j][r];
      }
    }
}

// ---------------------------------------------------------------------------
// Support kernels
// ---------------------------------------------------------------------------
__global__ void cvt_pad_f16_kernel(const float* __restrict__ src, _Float16* __restrict__ dst,
                                   int rows, int cols, int colsp) {
  size_t t = (size_t)blockIdx.x * blockDim.x + threadIdx.x;
  if (t >= (size_t)rows * colsp) return;
  int r = (int)(t / colsp), c = (int)(t % colsp);
  dst[t] = (c < cols) ? (_Float16)src[(size_t)r * cols + c] : (_Float16)0.f;
}

__global__ void transpose_in_kernel(const float* __restrict__ x, float* __restrict__ X0,
                                    int Bn, int Nn) {
  size_t t = (size_t)blockIdx.x * blockDim.x + threadIdx.x;
  size_t tot = (size_t)Bn * Nn * 3;
  if (t >= tot) return;
  int d = (int)(t % 3);
  size_t bn = t / 3;
  X0[(size_t)d * ((size_t)Bn * Nn) + bn] = x[t];
}

__global__ void knn16_kernel(const float* __restrict__ x, int* __restrict__ idx,
                             int Bn, int Nn) {
  int p = blockIdx.x * blockDim.x + threadIdx.x;
  if (p >= Bn * Nn) return;
  int b = p / Nn, n = p % Nn;
  const float* xb = x + (size_t)b * Nn * 3;
  float qx = xb[n * 3], qy = xb[n * 3 + 1], qz = xb[n * 3 + 2];
  float bd[16]; int bi[16];
  for (int i = 0; i < 16; ++i) { bd[i] = 3.4e38f; bi[i] = 0; }
  for (int m = 0; m < Nn; ++m) {
    float dx = xb[m * 3] - qx, dy = xb[m * 3 + 1] - qy, dz = xb[m * 3 + 2] - qz;
    float d = dx * dx + dy * dy + dz * dz;
    if (d < bd[15]) {
      int j = 15;
      while (j > 0 && bd[j - 1] > d) { bd[j] = bd[j - 1]; bi[j] = bi[j - 1]; --j; }
      bd[j] = d; bi[j] = m;
    }
  }
  for (int i = 0; i < 16; ++i) idx[(size_t)p * 16 + i] = bi[i];
}

// edge features: Ef[pos][Kp]  (pos = (b*N+n)*16+k), ch 0..C-1 = nbr-ctr, C..2C-1 = ctr
__global__ void edgefeat_kernel(const float* __restrict__ X, const int* __restrict__ idx,
                                _Float16* __restrict__ Ef, int C, int Bn, int Nn, int Kp) {
  size_t t = (size_t)blockIdx.x * blockDim.x + threadIdx.x;
  size_t total = (size_t)Bn * Nn * 16 * C;
  if (t >= total) return;
  int c = (int)(t % C);
  size_t pos = t / C;
  size_t bn = pos >> 4;
  int b = (int)(bn / Nn);
  int nb = idx[pos];
  size_t BNp = (size_t)Bn * Nn;
  float ctr = X[(size_t)c * BNp + bn];
  float nbr = X[(size_t)c * BNp + (size_t)b * Nn + nb];
  _Float16* o = Ef + pos * (size_t)Kp;
  o[c]     = (_Float16)(nbr - ctr);
  o[C + c] = (_Float16)ctr;
  if (c == 0) for (int z = 2 * C; z < Kp; ++z) o[z] = (_Float16)0.f;
}

__global__ void bn_stats_kernel(const float* __restrict__ Y, float* __restrict__ stats, int P) {
  int c = blockIdx.x;
  const float* row = Y + (size_t)c * P;
  float s = 0.f, s2 = 0.f;
  for (int i = threadIdx.x; i < P; i += blockDim.x) { float v = row[i]; s += v; s2 += v * v; }
  __shared__ float sh[256], sh2[256];
  sh[threadIdx.x] = s; sh2[threadIdx.x] = s2;
  __syncthreads();
  for (int st = 128; st > 0; st >>= 1) {
    if (threadIdx.x < st) { sh[threadIdx.x] += sh[threadIdx.x + st]; sh2[threadIdx.x] += sh2[threadIdx.x + st]; }
    __syncthreads();
  }
  if (threadIdx.x == 0) {
    float m = sh[0] / (float)P;
    float v = sh2[0] / (float)P - m * m;
    stats[2 * c] = m;
    stats[2 * c + 1] = rsqrtf(v + 1e-5f);
  }
}

// BN(train) + LeakyReLU(0.2) + optional max over kpool trailing positions.
// f32 result goes to Xo[C][P]; optional f16 copy goes position-major into
// XhT[p][ldxh] at channel offset coff (feeds the next NT GEMM directly).
__global__ void bn_act_kernel(const float* __restrict__ Y, const float* __restrict__ stats,
                              float* __restrict__ Xo, _Float16* __restrict__ XhT,
                              int ldxh, int coff, int C, int P, int kpool) {
  size_t t = (size_t)blockIdx.x * blockDim.x + threadIdx.x;
  if (t >= (size_t)C * P) return;
  int c = (int)(t / P), p = (int)(t % P);
  float mean = stats[2 * c], inv = stats[2 * c + 1];
  const float* y = Y + ((size_t)c * P + p) * (size_t)kpool;
  float mx = -3.4e38f;
  for (int k = 0; k < kpool; ++k) {
    float v = (y[k] - mean) * inv;
    v = v > 0.f ? v : 0.2f * v;
    mx = fmaxf(mx, v);
  }
  if (Xo)  Xo[t] = mx;
  if (XhT) XhT[(size_t)p * ldxh + coff + c] = (_Float16)mx;
}

__global__ void pool_kernel(const float* __restrict__ H, float* __restrict__ g,
                            int C, int Nn, int Bc) {
  int c = blockIdx.x, b = blockIdx.y;
  const float* row = H + (size_t)c * ((size_t)Bc * Nn) + (size_t)b * Nn;
  float mx = -3.4e38f, s = 0.f;
  for (int i = threadIdx.x; i < Nn; i += blockDim.x) { float v = row[i]; mx = fmaxf(mx, v); s += v; }
  __shared__ float shm[256], shs[256];
  shm[threadIdx.x] = mx; shs[threadIdx.x] = s;
  __syncthreads();
  for (int st = 128; st > 0; st >>= 1) {
    if (threadIdx.x < st) { shm[threadIdx.x] = fmaxf(shm[threadIdx.x], shm[threadIdx.x + st]); shs[threadIdx.x] += shs[threadIdx.x + st]; }
    __syncthreads();
  }
  if (threadIdx.x == 0) {
    g[(size_t)b * 2 * C + c]     = shm[0];
    g[(size_t)b * 2 * C + C + c] = shs[0] / (float)Nn;
  }
}

__global__ void bn_batch_lrelu_kernel(const float* __restrict__ D, float* __restrict__ outf,
                                      _Float16* __restrict__ outh, int O) {
  int o = blockIdx.x * blockDim.x + threadIdx.x;
  if (o >= O) return;
  float v[4];
  for (int b = 0; b < 4; ++b) v[b] = D[(size_t)o * 4 + b];
  float m = 0.25f * (v[0] + v[1] + v[2] + v[3]);
  float var = 0.f;
  for (int b = 0; b < 4; ++b) { float d = v[b] - m; var += d * d; }
  float inv = rsqrtf(0.25f * var + 1e-5f);
  for (int b = 0; b < 4; ++b) {
    float u = (v[b] - m) * inv;
    u = u > 0.f ? u : 0.2f * u;
    outf[(size_t)b * O + o] = u;
    outh[(size_t)b * O + o] = (_Float16)u;
  }
}

// D[O][4] + bias -> (optional relu) -> outf/outh row-major [4][O]
__global__ void bias_act_t_kernel(const float* __restrict__ D, const float* __restrict__ bias,
                                  float* __restrict__ outf, _Float16* __restrict__ outh,
                                  int O, int relu) {
  int t = blockIdx.x * blockDim.x + threadIdx.x;
  if (t >= 4 * O) return;
  int o = t >> 2, b = t & 3;
  float v = D[(size_t)o * 4 + b];
  if (bias) v += bias[o];
  if (relu) v = fmaxf(v, 0.f);
  if (outf) outf[(size_t)b * O + o] = v;
  if (outh) outh[(size_t)b * O + o] = (_Float16)v;
}

// out[b][o][n] = act(sum_c W[o][c]*in[b][c][n] + bias[o])
__global__ void conv1d_kernel(const float* __restrict__ in, const float* __restrict__ W,
                              const float* __restrict__ bias, float* __restrict__ out,
                              int Bc, int Cin, int Cout, int Nn, int relu) {
  size_t t = (size_t)blockIdx.x * blockDim.x + threadIdx.x;
  if (t >= (size_t)Bc * Cout * Nn) return;
  int n = (int)(t % Nn);
  int o = (int)((t / Nn) % Cout);
  int b = (int)(t / ((size_t)Nn * Cout));
  float s = bias[o];
  const float* ib = in + (size_t)b * Cin * Nn + n;
  const float* wr = W + (size_t)o * Cin;
  for (int c = 0; c < Cin; ++c) s += wr[c] * ib[(size_t)c * Nn];
  if (relu) s = fmaxf(s, 0.f);
  out[t] = s;
}

__global__ void knn_small_kernel(const float* __restrict__ pts, int* __restrict__ idx,
                                 int Bc, int Nn, int k) {
  int t = blockIdx.x * blockDim.x + threadIdx.x;
  if (t >= Bc * Nn) return;
  int b = t / Nn, n = t % Nn;
  const float* p = pts + (size_t)b * 3 * Nn;
  float qx = p[n], qy = p[Nn + n], qz = p[2 * Nn + n];
  float bd[3] = {3.4e38f, 3.4e38f, 3.4e38f};
  int bi[3] = {0, 0, 0};
  for (int m = 0; m < Nn; ++m) {
    float dx = p[m] - qx, dy = p[Nn + m] - qy, dz = p[2 * Nn + m] - qz;
    float d = dx * dx + dy * dy + dz * dz;
    if (d < bd[k - 1]) {
      int j = k - 1;
      while (j > 0 && bd[j - 1] > d) { bd[j] = bd[j - 1]; bi[j] = bi[j - 1]; --j; }
      bd[j] = d; bi[j] = m;
    }
  }
  for (int j = 0; j < k; ++j) idx[(size_t)t * k + j] = bi[j];
}

__global__ void transpose_pc_kernel(const float* __restrict__ pc, float* __restrict__ out,
                                    int Bc, int Nn) {
  int t = blockIdx.x * blockDim.x + threadIdx.x;
  if (t >= Bc * Nn * 3) return;
  int d = t % 3, n = (t / 3) % Nn, b = t / (3 * Nn);
  out[t] = pc[(size_t)b * 3 * Nn + (size_t)d * Nn + n];
}

// stage1: k=1 (self) mirror => out_mid[b][2j+s][d] = pc1[b][d][j] + pc2xyz[b][d][2j+s]
__global__ void expand1_kernel(const float* __restrict__ pc1, const float* __restrict__ pc2xyz,
                               float* __restrict__ out, int Bc) {
  int t = blockIdx.x * blockDim.x + threadIdx.x;
  if (t >= Bc * 3 * 64) return;
  int j = t % 64, d = (t / 64) % 3, b = t / 192;
  float c = pc1[(size_t)b * 192 + d * 64 + j];
  for (int s = 0; s < 2; ++s) {
    int n = 2 * j + s;
    out[((size_t)b * 128 + n) * 3 + d] = c + pc2xyz[((size_t)b * 3 + d) * 128 + n];
  }
}

// stage2: k=3 mirror => out_high[b][4j+s][d] = (2c - nbr_s) + pc3xyz ; s=3 -> c + pc3xyz
__global__ void expand2_kernel(const float* __restrict__ pc2xyz, const float* __restrict__ pc3xyz,
                               const int* __restrict__ idx2, float* __restrict__ out, int Bc) {
  int t = blockIdx.x * blockDim.x + threadIdx.x;
  if (t >= Bc * 3 * 128) return;
  int j = t % 128, d = (t / 128) % 3, b = t / 384;
  const float* pb = pc2xyz + (size_t)b * 384;
  float c = pb[d * 128 + j];
  for (int s = 0; s < 3; ++s) {
    int nb = idx2[((size_t)b * 128 + j) * 3 + s];
    float nv = pb[d * 128 + nb];
    int n = 4 * j + s;
    out[((size_t)b * 512 + n) * 3 + d] = 2.f * c - nv + pc3xyz[((size_t)b * 3 + d) * 512 + n];
  }
  int n = 4 * j + 3;
  out[((size_t)b * 512 + n) * 3 + d] = c + pc3xyz[((size_t)b * 3 + d) * 512 + n];
}

// ---------------------------------------------------------------------------
// Orchestration
// ---------------------------------------------------------------------------
extern "C" void kernel_launch(void* const* d_in, const int* in_sizes, int n_in,
                              void* d_out, int out_size, void* d_ws, size_t ws_size,
                              hipStream_t stream) {
  const int B = 4, N = 2048;
  const size_t BNp = (size_t)B * N;   // 8192
  const size_t P2  = BNp * 16;        // 131072
  const float* x = (const float*)d_in[0];

  char* base = (char*)d_ws;
  size_t cur = 0;
  auto alloc = [&](size_t bytes) -> char* {
    char* p = base + cur;
    cur += (bytes + 255) & ~(size_t)255;
    return p;
  };
  auto grd = [](size_t n) { return dim3((unsigned)((n + 255) / 256)); };

  auto cvt = [&](int i, int rows, int cols, int colsp) -> _Float16* {
    _Float16* dst = (_Float16*)alloc((size_t)rows * colsp * sizeof(_Float16));
    size_t total = (size_t)rows * colsp;
    cvt_pad_f16_kernel<<<grd(total), 256, 0, stream>>>((const float*)d_in[i], dst, rows, cols, colsp);
    return dst;
  };

  // f16 weights (cols padded to multiple of 8 where needed)
  _Float16* w1   = cvt(1, 64, 6, 8);
  _Float16* w2   = cvt(2, 64, 128, 128);
  _Float16* w21  = cvt(3, 128, 128, 128);
  _Float16* w3   = cvt(4, 128, 256, 256);
  _Float16* w31  = cvt(5, 256, 256, 256);
  _Float16* w4   = cvt(6, 512, 512, 512);
  _Float16* w5   = cvt(7, 1024, 768, 768);
  _Float16* wl1  = cvt(8, 2048, 2048, 2048);
  _Float16* wf1  = cvt(9, 1024, 2048, 2048);
  _Float16* wf2  = cvt(11, 512, 1024, 1024);
  _Float16* wf3  = cvt(13, 256, 512, 512);
  _Float16* wf11 = cvt(15, 65536, 1024, 1024);
  _Float16* wf21 = cvt(17, 8192, 512, 512);
  _Float16* wf31 = cvt(19, 192, 256, 256);

  int*       idx0 = (int*)alloc(P2 * sizeof(int));
  float*     X0   = (float*)alloc(3 * BNp * 4);
  float*     X    = (float*)alloc(768 * BNp * 4);        // x1|x2|x3|x4 stacked rows (f32)
  _Float16*  XhT  = (_Float16*)alloc(BNp * 768 * 2);     // position-major f16 concat
  float*     M1   = (float*)alloc(128 * BNp * 4);
  float*     M2   = (float*)alloc(256 * BNp * 4);
  float*     H5   = (float*)alloc(1024 * BNp * 4);
  _Float16*  Ef   = (_Float16*)alloc(P2 * 512 * 2);      // max 2C = 512
  float*     Y    = (float*)alloc((size_t)512 * P2 * 4); // max conv output
  float*     st   = (float*)alloc(2 * 1024 * 4);
  float*     g    = (float*)alloc(4 * 2048 * 4);
  _Float16*  gh   = (_Float16*)alloc(4 * 2048 * 2);
  float*     Dfc  = (float*)alloc((size_t)65536 * 4 * 4);
  float*     lat  = (float*)alloc(4 * 2048 * 4);
  _Float16*  lath = (_Float16*)alloc(4 * 2048 * 2);
  float*     x1f  = (float*)alloc(4 * 1024 * 4); _Float16* x1h = (_Float16*)alloc(4 * 1024 * 2);
  float*     x2f  = (float*)alloc(4 * 512 * 4);  _Float16* x2h = (_Float16*)alloc(4 * 512 * 2);
  float*     x3f  = (float*)alloc(4 * 256 * 4);  _Float16* x3h = (_Float16*)alloc(4 * 256 * 2);
  float*     pc1  = (float*)alloc(4 * 192 * 4);
  float*     pc2f = (float*)alloc(4 * 8192 * 4);
  float*     pc3f = (float*)alloc((size_t)4 * 65536 * 4);
  float*     pc3f2= (float*)alloc((size_t)4 * 64 * 512 * 4);
  float*     p2xy = (float*)alloc(4 * 3 * 128 * 4);
  float*     p3xy = (float*)alloc(4 * 3 * 512 * 4);
  int*       idx2 = (int*)alloc(4 * 128 * 3 * sizeof(int));

  auto gemm = [&](const _Float16* A, const _Float16* Bm, float* D, int M, int Np, int K, int ldb) {
    dim3 grid((Np + TBN - 1) / TBN, (M + TBM - 1) / TBM);
    gemm_f16_wmma<<<grid, dim3(WG), 0, stream>>>(A, Bm, D, M, Np, K, ldb);
  };
  auto stats = [&](const float* Yp, int C, int P) {
    bn_stats_kernel<<<dim3(C), 256, 0, stream>>>(Yp, st, P);
  };
  auto bnact = [&](const float* Yp, float* Xo, _Float16* Xhp, int coff, int C, int P, int kpool) {
    bn_act_kernel<<<grd((size_t)C * P), 256, 0, stream>>>(Yp, st, Xo, Xhp, 768, coff, C, P, kpool);
  };
  auto edgef = [&](const float* Xin, int C, int Kp) {
    edgefeat_kernel<<<grd(P2 * (size_t)C), 256, 0, stream>>>(Xin, idx0, Ef, C, B, N, Kp);
  };

  // ---- graph feature backbone ----
  transpose_in_kernel<<<grd(BNp * 3), 256, 0, stream>>>(x, X0, B, N);
  knn16_kernel<<<grd(BNp), 256, 0, stream>>>(x, idx0, B, N);

  edgef(X0, 3, 8);                                       // conv1
  gemm(w1, Ef, Y, 64, (int)P2, 8, 8);
  stats(Y, 64, (int)P2);
  bnact(Y, X, XhT, 0, 64, (int)BNp, 16);                 // x1 -> XhT[:, 0:64]

  edgef(X, 64, 128);                                     // conv2
  gemm(w2, Ef, Y, 64, (int)P2, 128, 128);
  stats(Y, 64, (int)P2);
  bnact(Y, X + 64 * BNp, XhT, 64, 64, (int)BNp, 16);     // x2 -> XhT[:, 64:128]

  gemm(w21, XhT, Y, 128, (int)BNp, 128, 768);            // conv2_1 on concat(x1,x2)
  stats(Y, 128, (int)BNp);
  bnact(Y, M1, nullptr, 0, 128, (int)BNp, 1);

  edgef(M1, 128, 256);                                   // conv3
  gemm(w3, Ef, Y, 128, (int)P2, 256, 256);
  stats(Y, 128, (int)P2);
  bnact(Y, X + 128 * BNp, XhT, 128, 128, (int)BNp, 16);  // x3 -> XhT[:, 128:256]

  gemm(w31, XhT, Y, 256, (int)BNp, 256, 768);            // conv3_1 on concat(x1..x3)
  stats(Y, 256, (int)BNp);
  bnact(Y, M2, nullptr, 0, 256, (int)BNp, 1);

  edgef(M2, 256, 512);                                   // conv4 (dominant GEMM)
  gemm(w4, Ef, Y, 512, (int)P2, 512, 512);
  stats(Y, 512, (int)P2);
  bnact(Y, X + 256 * BNp, XhT, 256, 512, (int)BNp, 16);  // x4 -> XhT[:, 256:768]

  gemm(w5, XhT, Y, 1024, (int)BNp, 768, 768);            // conv5 on concat(x1..x4)
  stats(Y, 1024, (int)BNp);
  bnact(Y, H5, nullptr, 0, 1024, (int)BNp, 1);

  pool_kernel<<<dim3(1024, 4), 256, 0, stream>>>(H5, g, 1024, N, B);
  cvt_pad_f16_kernel<<<grd(4 * 2048), 256, 0, stream>>>(g, gh, 4, 2048, 2048);

  // ---- FC decoder (all WMMA GEMMs with Np=4) ----
  gemm(wl1, gh, Dfc, 2048, 4, 2048, 2048);
  bn_batch_lrelu_kernel<<<grd(2048), 256, 0, stream>>>(Dfc, lat, lath, 2048);

  gemm(wf1, lath, Dfc, 1024, 4, 2048, 2048);
  bias_act_t_kernel<<<grd(4 * 1024), 256, 0, stream>>>(Dfc, (const float*)d_in[10], x1f, x1h, 1024, 1);
  gemm(wf2, x1h, Dfc, 512, 4, 1024, 1024);
  bias_act_t_kernel<<<grd(4 * 512), 256, 0, stream>>>(Dfc, (const float*)d_in[12], x2f, x2h, 512, 1);
  gemm(wf3, x2h, Dfc, 256, 4, 512, 512);
  bias_act_t_kernel<<<grd(4 * 256), 256, 0, stream>>>(Dfc, (const float*)d_in[14], x3f, x3h, 256, 1);

  gemm(wf31, x3h, Dfc, 192, 4, 256, 256);                // pc1_xyz (B,3,64)
  bias_act_t_kernel<<<grd(4 * 192), 256, 0, stream>>>(Dfc, (const float*)d_in[20], pc1, nullptr, 192, 0);
  gemm(wf21, x2h, Dfc, 8192, 4, 512, 512);               // pc2_feat (B,64,128)
  bias_act_t_kernel<<<grd(4 * 8192), 256, 0, stream>>>(Dfc, (const float*)d_in[18], pc2f, nullptr, 8192, 1);
  gemm(wf11, x1h, Dfc, 65536, 4, 1024, 1024);            // pc3_feat (B,128,512)
  bias_act_t_kernel<<<grd((size_t)4 * 65536), 256, 0, stream>>>(Dfc, (const float*)d_in[16], pc3f, nullptr, 65536, 1);

  conv1d_kernel<<<grd(4 * 3 * 128), 256, 0, stream>>>(pc2f, (const float*)d_in[25], (const float*)d_in[26], p2xy, 4, 64, 3, 128, 0);
  conv1d_kernel<<<grd((size_t)4 * 64 * 512), 256, 0, stream>>>(pc3f, (const float*)d_in[21], (const float*)d_in[22], pc3f2, 4, 128, 64, 512, 1);
  conv1d_kernel<<<grd(4 * 3 * 512), 256, 0, stream>>>(pc3f2, (const float*)d_in[23], (const float*)d_in[24], p3xy, 4, 64, 3, 512, 0);

  // ---- outputs: pc_low | pc_middle | pc_high ----
  float* out = (float*)d_out;
  transpose_pc_kernel<<<grd(4 * 64 * 3), 256, 0, stream>>>(pc1, out, 4, 64);
  expand1_kernel<<<grd(4 * 3 * 64), 256, 0, stream>>>(pc1, p2xy, out + 768, 4);
  knn_small_kernel<<<grd(4 * 128), 256, 0, stream>>>(p2xy, idx2, 4, 128, 3);
  expand2_kernel<<<grd(4 * 3 * 128), 256, 0, stream>>>(p2xy, p3xy, idx2, out + 2304, 4);
}
